// FFTMLP_86904368267649
// MI455X (gfx1250) — matrix operand order repaired and checked by reading, not compiled, not run on hardware
//
#include <hip/hip_runtime.h>
#include <stdint.h>

// ---------------------------------------------------------------------------
// FFT-MLP energies == GEMM: out[16384,1024] = Xflat[16384,4098] * W^T[4098,1024]
//   Xflat = x viewed flat per row (real channel then imag channel, contiguous)
//   W[o, 0:2049]    = fr[o,:] + fi[o,:]
//   W[o, 2049:4098] = fr[o,:] - fi[o,:]
// Strategy: bf16 WMMA (v_wmma_f32_16x16x32_bf16), f32 accumulate.
//   - W tile staged via global_load_async_to_lds_b128 (ASYNCcnt pipeline)
//   - A (x) read as f32 once, converted with v_cvt_pk_bf16_f32 on the fly
// ---------------------------------------------------------------------------

typedef __attribute__((ext_vector_type(16))) __bf16 v16bf;
typedef __attribute__((ext_vector_type(2)))  __bf16 v2bf;
typedef __attribute__((ext_vector_type(8)))  float  v8f;
typedef __attribute__((ext_vector_type(2)))  float  vf2;

#define BATCH   16384
#define ODIM    1024
#define FREQ    2049
#define KTOT    4098            // 2 * 2049
#define KPAD    4160            // 130 * 32
#define KSTEP   32
#define NSTAGES (KPAD / KSTEP)  // 130
#define TM      128             // block tile M
#define TN      256             // block tile N
#define LDA     40              // padded LDS row pitch (bf16 elems): 32 + 8
#define LDW     40

__device__ __forceinline__ unsigned short f2bf(float f) {
  union { float f; uint32_t u; } c; c.f = f;
  uint32_t u = c.u;
  return (unsigned short)((u + 0x7FFFu + ((u >> 16) & 1u)) >> 16);
}

__device__ __forceinline__ uint32_t pk_bf16(float a, float b) {
  union { v2bf v; uint32_t u; } cv;
  vf2 t; t.x = a; t.y = b;
  cv.v = __builtin_convertvector(t, v2bf);   // -> v_cvt_pk_bf16_f32
  return cv.u;
}

__device__ __forceinline__ void wait_async0() {
#if __has_builtin(__builtin_amdgcn_s_wait_asynccnt)
  __builtin_amdgcn_s_wait_asynccnt(0);
#else
  asm volatile("s_wait_asynccnt 0x0" ::: "memory");
#endif
}

// Build padded bf16 weight matrix Wb[ODIM][KPAD] in workspace.
__global__ void fftmlp_prep(const float* __restrict__ fr,
                            const float* __restrict__ fi,
                            unsigned short* __restrict__ wb) {
  int idx = blockIdx.x * 256 + threadIdx.x;
  if (idx >= ODIM * KPAD) return;
  int o = idx / KPAD;
  int k = idx - o * KPAD;
  float v = 0.0f;
  if (k < FREQ) {
    v = fr[o * FREQ + k] + fi[o * FREQ + k];
  } else if (k < KTOT) {
    int kk = k - FREQ;
    v = fr[o * FREQ + kk] - fi[o * FREQ + kk];
  }
  wb[idx] = f2bf(v);
}

__global__ __launch_bounds__(256) void fftmlp_gemm(
    const float* __restrict__ x,            // [BATCH, KTOT] (flat view)
    const unsigned short* __restrict__ wb,  // [ODIM, KPAD] bf16
    float* __restrict__ out) {              // [BATCH, ODIM]
  __shared__ unsigned short ldsA[2][TM * LDA];
  __shared__ unsigned short ldsW[2][TN * LDW];

  const int tid  = threadIdx.x;
  const int lane = tid & 31;
  const int wv   = tid >> 5;     // wave 0..7
  const int wm   = wv & 1;       // 2 waves along M
  const int wn   = wv >> 1;      // 4 waves along N
  const int l15  = lane & 15;
  const int kh   = lane >> 4;    // K-half selector per WMMA layout

  const int mtile = blockIdx.y * TM;
  const int ntile = blockIdx.x * TN;

  v8f acc[4][4];
#pragma unroll
  for (int i = 0; i < 4; ++i)
#pragma unroll
    for (int j = 0; j < 4; ++j)
      acc[i][j] = (v8f){0.f, 0.f, 0.f, 0.f, 0.f, 0.f, 0.f, 0.f};

  float2 aReg[8];   // staged A (f32, converted to bf16 at LDS store)

  // --- W tile: async memory -> LDS, no VGPR data round-trip (ASYNCcnt) ---
  auto asyncLoadW = [&](int buf, int ks) {
    const unsigned short* gbase = wb + (size_t)ntile * KPAD + ks;
#pragma unroll
    for (int i = 0; i < 4; ++i) {
      int c = i * 256 + tid;            // 1024 b128 chunks: 256 rows x 4
      int row = c >> 2, j = c & 3;
      unsigned gofs = (unsigned)((row * KPAD + j * 8) * 2);           // bytes
      unsigned lofs = (unsigned)(uintptr_t)&ldsW[buf][row * LDW + j * 8];
      asm volatile("global_load_async_to_lds_b128 %0, %1, %2"
                   :: "v"(lofs), "v"(gofs), "s"(gbase)
                   : "memory");
    }
  };

  // --- A tile: global f32 -> regs (float2 granular; KTOT even) ---
  auto loadA = [&](int ks) {
#pragma unroll
    for (int i = 0; i < 8; ++i) {
      int c = i * 256 + tid;            // 2048 float2 chunks: 128 rows x 16
      int row = c >> 4, j = c & 15;
      int gk = ks + j * 2;
      if (gk < KTOT) {
        aReg[i] = *(const float2*)(x + (size_t)(mtile + row) * KTOT + gk);
      } else {
        aReg[i] = make_float2(0.f, 0.f);
      }
    }
  };

  // --- A tile: regs -> LDS with packed bf16 conversion ---
  auto storeA = [&](int buf) {
#pragma unroll
    for (int i = 0; i < 8; ++i) {
      int c = i * 256 + tid;
      int row = c >> 4, j = c & 15;
      *(uint32_t*)&ldsA[buf][row * LDA + j * 2] = pk_bf16(aReg[i].x, aReg[i].y);
    }
  };

  auto compute = [&](int buf) {
    union BF { v16bf v; uint4 q[2]; };
    const unsigned short* aBase = &ldsA[buf][(wm * 64 + l15) * LDA + kh * 8];
    BF bfr[4];
    BF acur, anext;
    // First A fragment, then all B fragments (latency overlaps)
    acur.q[0] = *(const uint4*)(aBase);
    acur.q[1] = *(const uint4*)(aBase + 16);
#pragma unroll
    for (int in = 0; in < 4; ++in) {
      const unsigned short* p =
          &ldsW[buf][(wn * 64 + in * 16 + l15) * LDW + kh * 8];
      bfr[in].q[0] = *(const uint4*)(p);
      bfr[in].q[1] = *(const uint4*)(p + 16);
    }
#pragma unroll
    for (int im = 0; im < 4; ++im) {
      if (im < 3) {  // prefetch next A fragment under this WMMA group
        const unsigned short* p = aBase + (im + 1) * 16 * LDA;
        anext.q[0] = *(const uint4*)(p);
        anext.q[1] = *(const uint4*)(p + 16);
      }
#pragma unroll
      for (int in = 0; in < 4; ++in) {
        acc[im][in] = __builtin_amdgcn_wmma_f32_16x16x32_bf16(
            false, acur.v, false, bfr[in].v, (short)0, acc[im][in],
            false, false);
      }
      acur = anext;
    }
  };

  // Prologue: stage 0
  asyncLoadW(0, 0);
  loadA(0);
  storeA(0);
  wait_async0();
  __syncthreads();

  for (int s = 0; s < NSTAGES; ++s) {
    int buf = s & 1;
    if (s + 1 < NSTAGES) {
      asyncLoadW(buf ^ 1, (s + 1) * KSTEP);  // async W for next stage
      loadA((s + 1) * KSTEP);                // global A -> regs
    }
    compute(buf);                            // WMMA on current buffers
    if (s + 1 < NSTAGES) storeA(buf ^ 1);    // regs -> LDS (cvt_pk_bf16)
    wait_async0();                           // drain ASYNCcnt before barrier
    __syncthreads();
  }

  // D layout: VGPR r -> M = r + 8*kh, N = l15
#pragma unroll
  for (int im = 0; im < 4; ++im) {
#pragma unroll
    for (int in = 0; in < 4; ++in) {
      int gm0 = mtile + wm * 64 + im * 16 + kh * 8;
      int gn  = ntile + wn * 64 + in * 16 + l15;
#pragma unroll
      for (int r = 0; r < 8; ++r) {
        out[(size_t)(gm0 + r) * ODIM + gn] = acc[im][in][r];
      }
    }
  }
}

extern "C" void kernel_launch(void* const* d_in, const int* in_sizes, int n_in,
                              void* d_out, int out_size, void* d_ws, size_t ws_size,
                              hipStream_t stream) {
  const float* x  = (const float*)d_in[0];    // [16384, 2, 2049] f32
  const float* fr = (const float*)d_in[1];    // [1024, 2049] f32
  const float* fi = (const float*)d_in[2];    // [1024, 2049] f32
  float* out = (float*)d_out;                 // [16384, 1024] f32
  unsigned short* wb = (unsigned short*)d_ws; // [1024, 4160] bf16 (~8.1 MB)

  int total = ODIM * KPAD;
  fftmlp_prep<<<(total + 255) / 256, 256, 0, stream>>>(fr, fi, wb);

  dim3 grid(ODIM / TN, BATCH / TM);  // (4, 128): N fastest for L2 reuse of A
  fftmlp_gemm<<<grid, 256, 0, stream>>>(x, wb, out);
}
